// VectorQuantizer_12781822673326
// MI455X (gfx1250) — matrix operand order repaired and checked by reading, not compiled, not run on hardware
//
#include <hip/hip_runtime.h>

typedef __attribute__((ext_vector_type(16))) _Float16 v16h;
typedef __attribute__((ext_vector_type(8)))  _Float16 v8h;
typedef __attribute__((ext_vector_type(8)))  float    v8f;
typedef __attribute__((ext_vector_type(4)))  unsigned int u32x4;
typedef __attribute__((ext_vector_type(8)))  int      i32x8;
typedef __attribute__((ext_vector_type(4)))  int      i32x4;

#define NUM_EMB   8192
#define EMB_DIM   256
#define NROWS     32768      // 32*32*32 spatial positions
#define OUT_ELEMS 8388608    // 32*256*32*32
#define EMB_SCALE 4096.0f
#define INV_EMB_SCALE (1.0f/4096.0f)
#define TILE_C    32         // codes staged per LDS tile
#define LSTR      264        // padded LDS row stride (halves): 256 + 4dw TDM pad

// ---------------------------------------------------------------------------
// Kernel 1: x [B,D,H,W] f32  ->  A [n=(b,h,w), d] f16  (LDS tile transpose)
// ---------------------------------------------------------------------------
__global__ __launch_bounds__(256) void vq_transpose_kernel(
    const float* __restrict__ x, _Float16* __restrict__ A)
{
    __shared__ float tile[32][33];
    const int gid = blockIdx.x;              // 8192 blocks = b(32) * h(32) * dt(8)
    const int dt = gid & 7, h = (gid >> 3) & 31, b = gid >> 8;
    const int tw = threadIdx.x & 31, tq = threadIdx.x >> 5;
#pragma unroll
    for (int r = 0; r < 4; ++r) {
        const int dl = tq * 4 + r;
        const int d  = dt * 32 + dl;
        tile[dl][tw] = x[(((size_t)b * 256 + d) * 32 + h) * 32 + tw]; // coalesced over w
    }
    __syncthreads();
#pragma unroll
    for (int r = 0; r < 4; ++r) {
        const int wl = tq * 4 + r;
        const int n  = b * 1024 + h * 32 + wl;
        A[(size_t)n * 256 + dt * 32 + tw] = (_Float16)tile[tw][wl];   // coalesced over d
    }
}

// ---------------------------------------------------------------------------
// Kernel 2: emb f32 -> emb f16 (scaled by 4096) + bias[c] = 0.5*||e_c||^2
// ---------------------------------------------------------------------------
__global__ __launch_bounds__(256) void vq_embprep_kernel(
    const float* __restrict__ emb, _Float16* __restrict__ embh,
    float* __restrict__ bias)
{
    const int wave = threadIdx.x >> 5, lane = threadIdx.x & 31;
    const int c = blockIdx.x * 8 + wave;     // 1024 blocks, one wave per code
    const float* src = emb  + (size_t)c * 256 + lane * 8;
    _Float16*    dst = embh + (size_t)c * 256 + lane * 8;
    float s = 0.f;
#pragma unroll
    for (int i = 0; i < 8; ++i) {
        const float e = src[i];
        s += e * e;
        dst[i] = (_Float16)(e * EMB_SCALE);
    }
#pragma unroll
    for (int m = 16; m >= 1; m >>= 1) s += __shfl_xor(s, m, 32);
    if (lane == 0) bias[c] = 0.5f * s;
}

// ---------------------------------------------------------------------------
// TDM: DMA one codebook tile (32 codes x 256 halves) global -> LDS, with
// hardware padding of 4 dwords after every 128 dwords (512B row) so LDS rows
// land on a 264-half stride (bank-conflict-free b128 reads).
// D# layout per CDNA5 ISA 8.3/8.4.  6-arg builtin (clang-23 form): the three
// trailing descriptor-group operands are unused for a 2-D tensor -> zeros.
// ---------------------------------------------------------------------------
__device__ __forceinline__ void tdm_load_tile(const _Float16* gsrc,
                                              unsigned lds_byte_addr)
{
    const unsigned long long ga = (unsigned long long)(uintptr_t)gsrc;
    u32x4 g0;
    g0[0] = 1u;                                   // count=1 (valid user descriptor)
    g0[1] = lds_byte_addr;                        // lds_addr [63:32]
    g0[2] = (unsigned)(ga & 0xffffffffu);         // global_addr[31:0]
    g0[3] = (unsigned)((ga >> 32) & 0x01ffffffu)  // global_addr[56:32]
          | 0x80000000u;                          // type=2 ("image") [127:126]
    i32x8 g1;
    g1[0] = (1 << 16)      // data_size = 2 bytes
          | (1 << 20)      // pad_enable
          | (6 << 22)      // pad_interval: 2<<6 = 128 dwords (512B) between pads
          | (3 << 25);     // pad_amount: 4 dwords (8 halves)
    g1[1] = (int)(256u << 16);   // tensor_dim0[15:0]=256 (bits 79:48 low half)
    g1[2] = (int)(32u  << 16);   // tensor_dim0[31:16]=0 | tensor_dim1[15:0]=32
    g1[3] = (int)(256u << 16);   // tensor_dim1[31:16]=0 | tile_dim0=256
    g1[4] = 32;                  // tile_dim1=32 | tile_dim2=0
    g1[5] = 256;                 // tensor_dim0_stride[31:0] = 256 elements
    g1[6] = 0;                   // stride0[47:32] | tensor_dim1_stride[15:0]
    g1[7] = 0;                   // tensor_dim1_stride[47:16]
    const i32x4 z4 = {0, 0, 0, 0};                // VADDR2/3 groups unused (<=2D)
    const i32x8 z8 = {0, 0, 0, 0, 0, 0, 0, 0};
    __builtin_amdgcn_tensor_load_to_lds(g0, g1, z4, z4, z8, 0);
}

__device__ __forceinline__ v16h ldB(const _Float16* bp, int kd, int hs)
{
    v8h lo = *(const v8h*)(bp + kd * 32 + hs * 8);
    v8h hi = *(const v8h*)(bp + kd * 32 + 16 + hs * 8);
    return __builtin_shufflevector(lo, hi,
               0,1,2,3,4,5,6,7,8,9,10,11,12,13,14,15);
}

// ---------------------------------------------------------------------------
// Kernel 3: fused  score = z.e - 0.5||e||^2  GEMM (WMMA f16) + row argmax.
// 256 blocks x 8 waves; wave owns 16 rows (A resident in 64 VGPRs), streams
// all 8192 codes through a TDM-fed double-buffered LDS stage of 32 codes.
// ---------------------------------------------------------------------------
__global__ __launch_bounds__(256) void vq_argmin_kernel(
    const _Float16* __restrict__ A, const _Float16* __restrict__ embh,
    const float* __restrict__ bias, int* __restrict__ idx,
    float* __restrict__ idxf)
{
    __shared__ _Float16 Bt[2][TILE_C * LSTR];     // 2 x 16.5KB
    const int tid  = threadIdx.x;
    const int wave = tid >> 5, lane = tid & 31;
    const int hs = lane >> 4, l15 = lane & 15;
    const int m_base = blockIdx.x * 128 + wave * 16;

    // --- A fragments: 16 rows x 256 d, resident in registers (8 x v16h) ---
    // ISA 16-bit A 16x32 layout: halves 0..7 -> d0+8*hs.. ; 8..15 -> d0+16+8*hs..
    v16h a[8];
    {
        const _Float16* rowp = A + (size_t)(m_base + l15) * 256;
#pragma unroll
        for (int kd = 0; kd < 8; ++kd) {
            v8h lo = *(const v8h*)(rowp + kd * 32 + hs * 8);
            v8h hi = *(const v8h*)(rowp + kd * 32 + 16 + hs * 8);
            a[kd] = __builtin_shufflevector(lo, hi,
                        0,1,2,3,4,5,6,7,8,9,10,11,12,13,14,15);
        }
    }

    float bestVal[16];
    int   bestIdx[16];
#pragma unroll
    for (int r = 0; r < 16; ++r) { bestVal[r] = -3.4e38f; bestIdx[r] = 0; }

    const unsigned lds_base  = (unsigned)(uintptr_t)&Bt[0][0];
    const unsigned buf_bytes = TILE_C * LSTR * 2;

    if (wave == 0) {                              // DMA tile 0
        tdm_load_tile(embh, lds_base);
        __builtin_amdgcn_s_wait_tensorcnt(0);
    }
    __syncthreads();

    const int NT = NUM_EMB / TILE_C;              // 256 tiles
    for (int t = 0; t < NT; ++t) {
        const int cur = t & 1;
        if (wave == 0 && t + 1 < NT)              // async DMA next tile
            tdm_load_tile(embh + (size_t)(t + 1) * TILE_C * EMB_DIM,
                          lds_base + (unsigned)(cur ^ 1) * buf_bytes);

        // B 32x16 layout: lane's column = l15; halves 0..7 -> k = 8*hs + h ; 8..15 -> k = 16 + 8*hs + (h-8)
        const _Float16* bp0 = &Bt[cur][l15 * LSTR];
        const _Float16* bp1 = &Bt[cur][(16 + l15) * LSTR];
        v8f acc0 = {}, acc1 = {};
        v16h b0 = ldB(bp0, 0, hs), b1 = ldB(bp1, 0, hs);
#pragma unroll
        for (int kd = 0; kd < 8; ++kd) {          // B loads pipelined 1 step ahead
            v16h n0 = b0, n1 = b1;
            if (kd < 7) { n0 = ldB(bp0, kd + 1, hs); n1 = ldB(bp1, kd + 1, hs); }
            acc0 = __builtin_amdgcn_wmma_f32_16x16x32_f16(
                       false, a[kd], false, b0, (short)0, acc0, false, false);
            acc1 = __builtin_amdgcn_wmma_f32_16x16x32_f16(
                       false, a[kd], false, b1, (short)0, acc1, false, false);
            b0 = n0; b1 = n1;
        }
        const int c0 = t * TILE_C + l15;
        const int c1 = c0 + 16;
        const float bv0 = bias[c0];
        const float bv1 = bias[c1];
#pragma unroll
        for (int r = 0; r < 8; ++r) {
            const float s0 = acc0[r] * INV_EMB_SCALE - bv0;
            if (s0 > bestVal[r])     { bestVal[r]     = s0; bestIdx[r]     = c0; }
            const float s1 = acc1[r] * INV_EMB_SCALE - bv1;
            if (s1 > bestVal[r + 8]) { bestVal[r + 8] = s1; bestIdx[r + 8] = c1; }
        }
        if (wave == 0) __builtin_amdgcn_s_wait_tensorcnt(0);
        __syncthreads();
    }

    // Merge the two code-halves per row (both live in the same lane).
#pragma unroll
    for (int r = 0; r < 8; ++r) {
        if (bestVal[r + 8] > bestVal[r] ||
            (bestVal[r + 8] == bestVal[r] && bestIdx[r + 8] < bestIdx[r])) {
            bestVal[r] = bestVal[r + 8]; bestIdx[r] = bestIdx[r + 8];
        }
    }
    // C-frag layout: lane holds column n=l15, VGPR r -> row r + 8*hs.
    // Reduce argmax across the 16 lanes sharing a row set (xor masks 1,2,4,8).
#pragma unroll
    for (int r = 0; r < 8; ++r) {
        float v = bestVal[r];
        int   bi = bestIdx[r];
#pragma unroll
        for (int m = 1; m < 16; m <<= 1) {
            const float ov = __shfl_xor(v, m, 32);
            const int   oi = __shfl_xor(bi, m, 32);
            if (ov > v || (ov == v && oi < bi)) { v = ov; bi = oi; }
        }
        bestIdx[r] = bi;
    }
    if (l15 == 0) {
        const int row0 = m_base + hs * 8;
#pragma unroll
        for (int r = 0; r < 8; ++r) {
            idx[row0 + r]  = bestIdx[r];
            idxf[row0 + r] = (float)bestIdx[r];
        }
    }
}

// ---------------------------------------------------------------------------
// Kernel 4: out[NCHW] = emb[idx]  (straight-through forward), SSE partials
// ---------------------------------------------------------------------------
__global__ __launch_bounds__(256) void vq_gather_kernel(
    const float* __restrict__ x, const float* __restrict__ emb,
    const int* __restrict__ idx, float* __restrict__ out,
    float* __restrict__ partial)
{
    const int tid = blockIdx.x * 256 + threadIdx.x;   // NCHW-linear -> coalesced x/out
    const int w = tid & 31, h = (tid >> 5) & 31;
    const int d = (tid >> 10) & 255, b = tid >> 18;
    const int n = b * 1024 + h * 32 + w;
    const int c = idx[n];
    const float e  = emb[(size_t)c * 256 + d];
    const float xv = x[tid];
    out[tid] = e;
    const float df = e - xv;
    float s = df * df;
#pragma unroll
    for (int m = 16; m >= 1; m >>= 1) s += __shfl_xor(s, m, 32);
    __shared__ float wsum[8];
    const int wave = threadIdx.x >> 5, lane = threadIdx.x & 31;
    if (lane == 0) wsum[wave] = s;
    __syncthreads();
    if (threadIdx.x == 0) {
        float tot = 0.f;
#pragma unroll
        for (int i = 0; i < 8; ++i) tot += wsum[i];
        partial[blockIdx.x] = tot;
    }
}

// ---------------------------------------------------------------------------
// Kernel 5: deterministic final loss reduction
// loss = (1 + 0.25) * mean((q - x)^2)
// ---------------------------------------------------------------------------
__global__ __launch_bounds__(256) void vq_loss_kernel(
    const float* __restrict__ partial, float* __restrict__ loss)
{
    float s = 0.f;
    for (int i = threadIdx.x; i < 32768; i += 256) s += partial[i];
#pragma unroll
    for (int m = 16; m >= 1; m >>= 1) s += __shfl_xor(s, m, 32);
    __shared__ float wsum[8];
    const int wave = threadIdx.x >> 5, lane = threadIdx.x & 31;
    if (lane == 0) wsum[wave] = s;
    __syncthreads();
    if (threadIdx.x == 0) {
        float tot = 0.f;
#pragma unroll
        for (int i = 0; i < 8; ++i) tot += wsum[i];
        *loss = 1.25f * tot / 8388608.0f;
    }
}

// ---------------------------------------------------------------------------
extern "C" void kernel_launch(void* const* d_in, const int* in_sizes, int n_in,
                              void* d_out, int out_size, void* d_ws, size_t ws_size,
                              hipStream_t stream)
{
    const float* x   = (const float*)d_in[0];   // [32,256,32,32]
    const float* emb = (const float*)d_in[1];   // [8192,256]

    float* out_f = (float*)d_out;               // out | loss | idx(as float)
    float* loss  = out_f + OUT_ELEMS;
    float* idxf  = out_f + OUT_ELEMS + 1;

    char* ws = (char*)d_ws;
    _Float16* A    = (_Float16*)ws; ws += (size_t)NROWS   * EMB_DIM * 2; // 16 MB
    _Float16* embh = (_Float16*)ws; ws += (size_t)NUM_EMB * EMB_DIM * 2; //  4 MB
    float* bias    = (float*)ws;    ws += (size_t)NUM_EMB * 4;
    int*   idx     = (int*)ws;      ws += (size_t)NROWS   * 4;
    float* partial = (float*)ws;    ws += (size_t)32768   * 4;

    vq_transpose_kernel<<<8192, 256, 0, stream>>>(x, A);
    vq_embprep_kernel  <<<1024, 256, 0, stream>>>(emb, embh, bias);
    vq_argmin_kernel   <<<256,  256, 0, stream>>>(A, embh, bias, idx, idxf);
    vq_gather_kernel   <<<32768,256, 0, stream>>>(x, emb, idx, out_f, partial);
    vq_loss_kernel     <<<1,    256, 0, stream>>>(partial, loss);
}
